// SSM_17377437680303
// MI455X (gfx1250) — matrix operand order repaired and checked by reading, not compiled, not run on hardware
//
#include <hip/hip_runtime.h>
#include <hip/hip_bf16.h>

// ---------------------------------------------------------------------------
// Problem constants (from reference)
// ---------------------------------------------------------------------------
#define BATCH   2
#define SEQLEN  2048
#define DMODEL  1024
#define DSTATE  16
#define DCONV   4

#define M_TOTAL (BATCH * SEQLEN)   // 4096 rows of the delta GEMM
#define K_DIM   DMODEL             // 1024
#define N_DIM   DMODEL             // 1024

#define NCHUNK  16
#define LCHUNK  (SEQLEN / NCHUNK)  // 128
#define NGRP    (BATCH * DMODEL)   // 2048 (b,d) channels

typedef unsigned short u16;
typedef __attribute__((ext_vector_type(16))) __bf16 v16bf;
typedef __attribute__((ext_vector_type(8)))  float  v8f;

union FragBF {
    uint4 q[2];   // two 16-byte chunks
    v16bf v;      // 8 VGPRs of packed bf16
};

// ---------------------------------------------------------------------------
// Kernel 1: fp32 -> bf16 (round to nearest even) for x and Wd
// ---------------------------------------------------------------------------
__global__ __launch_bounds__(256)
void cvt_bf16_kernel(const float* __restrict__ x,
                     const float* __restrict__ wd,
                     u16* __restrict__ xbf,
                     u16* __restrict__ wdbf)
{
    const int stride = gridDim.x * blockDim.x;
    const int tid = blockIdx.x * blockDim.x + threadIdx.x;
    const int total_x = M_TOTAL * K_DIM;
    const int total_w = N_DIM * K_DIM;
    for (int i = tid; i < total_x; i += stride) {
        unsigned u = __float_as_uint(x[i]);
        unsigned r = u + 0x7FFFu + ((u >> 16) & 1u);   // RNE
        xbf[i] = (u16)(r >> 16);
    }
    for (int i = tid; i < total_w; i += stride) {
        unsigned u = __float_as_uint(wd[i]);
        unsigned r = u + 0x7FFFu + ((u >> 16) & 1u);
        wdbf[i] = (u16)(r >> 16);
    }
}

// ---------------------------------------------------------------------------
// Kernel 2: delta = silu(x @ Wd^T + bd) via v_wmma_f32_16x16x32_bf16.
//
// Workgroup = 8 waves covering a 128(M) x 64(N) tile; all waves share the
// same 64-column B panel. Per 32-wide K step the panel (64 rows x 64 B)
// is staged into LDS with GLOBAL_LOAD_ASYNC_TO_LDS_B128 (ASYNCcnt path),
// double-buffered; the A fragment is software-pipelined in registers so no
// global-load wait sits in front of the WMMAs. Rows are padded to 80 B so
// half-wave ds_load_b128 reads are bank-conflict free.
// ---------------------------------------------------------------------------
#define BROW_U16 40                 // 32 bf16 payload + 8 bf16 pad = 80 bytes
#define NKSTEP   (K_DIM / 32)       // 32

__global__ __launch_bounds__(256)
void gemm_delta_kernel(const u16* __restrict__ Abf,   // M x K bf16 (x)
                       const u16* __restrict__ Bbf,   // N x K bf16 (Wd)
                       const float* __restrict__ bd,  // (N,)
                       float* __restrict__ delta)     // M x N f32
{
    __shared__ u16 ldsB[2][64 * BROW_U16];            // 2 x 5120 B

    const int tid  = threadIdx.x;
    const int lane = tid & 31;
    const int wave = tid >> 5;
    const int mblk = blockIdx.x >> 4;                 // 0..31
    const int n0   = (blockIdx.x & 15) * 64;
    const int m0   = mblk * 128 + wave * 16;

    const int row = lane & 15;                        // M-row (A) / N-col (B)
    const int hi  = lane >> 4;                        // half-wave selector

    // --- async staging assignment: one b128 per thread per K step ---------
    const int srow   = tid >> 2;                      // panel row 0..63
    const int schunk = tid & 3;                       // 16-byte chunk 0..3
    const unsigned lds_dst0 =
        (unsigned)(size_t)&ldsB[0][srow * BROW_U16] + (unsigned)(schunk * 16);
    const unsigned lds_dst1 =
        (unsigned)(size_t)&ldsB[1][srow * BROW_U16] + (unsigned)(schunk * 16);
    const unsigned gbase = (unsigned)((n0 + srow) * K_DIM * 2 + schunk * 16);

    v8f acc0 = {}; v8f acc1 = {}; v8f acc2 = {}; v8f acc3 = {};
    const u16* aRow = Abf + (size_t)(m0 + row) * K_DIM;

    // prologue: stage B panel for k0 = 0, prefetch A fragment for k0 = 0
    {
        unsigned voff = gbase;
        asm volatile("global_load_async_to_lds_b128 %0, %1, %2"
                     :: "v"(lds_dst0), "v"(voff), "s"(Bbf) : "memory");
    }
    FragBF a_cur, a_nxt;
    a_cur.q[0] = *(const uint4*)(aRow + hi * 8);
    a_cur.q[1] = *(const uint4*)(aRow + 16 + hi * 8);

    for (int i = 0; i < NKSTEP; ++i) {
        const int k0  = i * 32;
        const int cur = i & 1;

        asm volatile("s_wait_asynccnt 0x0" ::: "memory");
        __syncthreads();                              // panel `cur` ready

        if (i + 1 < NKSTEP) {
            // prefetch next B panel (async -> LDS) and next A frag (regs)
            unsigned voff = gbase + (unsigned)((k0 + 32) * 2);
            unsigned dst  = (cur ? lds_dst0 : lds_dst1);
            asm volatile("global_load_async_to_lds_b128 %0, %1, %2"
                         :: "v"(dst), "v"(voff), "s"(Bbf) : "memory");
            a_nxt.q[0] = *(const uint4*)(aRow + (k0 + 32) + hi * 8);
            a_nxt.q[1] = *(const uint4*)(aRow + (k0 + 48) + hi * 8);
        }

        #pragma unroll
        for (int t = 0; t < 4; ++t) {
            // B fragment from LDS: lane holds B[k0+hi*8+e, n0+t*16+row] (e<8), B[k0+16+hi*8+(e-8), ...] (e>=8)
            const uint4* bq = (const uint4*)
                &ldsB[cur][(unsigned)(t * 16 + row) * BROW_U16 + hi * 8];
            FragBF bf;
            bf.q[0] = bq[0];
            bf.q[1] = bq[2];

            v8f c = (t == 0) ? acc0 : (t == 1) ? acc1 : (t == 2) ? acc2 : acc3;
            c = __builtin_amdgcn_wmma_f32_16x16x32_bf16(
                    false, a_cur.v, false, bf.v, (short)0, c, false, false);
            if      (t == 0) acc0 = c;
            else if (t == 1) acc1 = c;
            else if (t == 2) acc2 = c;
            else             acc3 = c;
        }
        a_cur = a_nxt;
    }

    // Epilogue: bias + SiLU, store f32.
    // C/D layout: VGPR r, lane -> (m = r + hi*8, n = lane&15).
    #pragma unroll
    for (int t = 0; t < 4; ++t) {
        const v8f acc = (t == 0) ? acc0 : (t == 1) ? acc1 : (t == 2) ? acc2 : acc3;
        const int n = n0 + t * 16 + row;
        const float bias = bd[n];
        #pragma unroll
        for (int r = 0; r < 8; ++r) {
            const int m = m0 + r + hi * 8;
            float v = acc[r] + bias;
            float s = v / (1.0f + __expf(-v));   // silu
            delta[(size_t)m * N_DIM + n] = s;
        }
    }
}

// ---------------------------------------------------------------------------
// Chunked selective scan (3 passes). h' = Ad*h + b is linear-diagonal, so:
//   pass 1: per chunk (zero init) compute P = prod(Ad) and local end state.
//   pass 2: tiny sequential combine over the 16 chunks -> chunk entry states.
//   pass 3: redo local scans seeded with the true entry state; emit y.
// 16 lanes per (b,d) channel (one state per lane); causal conv is computed
// on the fly from a rolling register window.
// ---------------------------------------------------------------------------
__global__ __launch_bounds__(256)
void scan_part1_kernel(const float* __restrict__ x,
                       const float* __restrict__ Ap,
                       const float* __restrict__ Bp,
                       const float* __restrict__ conv_w,
                       const float* __restrict__ conv_b,
                       const float* __restrict__ delta,
                       float* __restrict__ Pbuf,     // (NCHUNK, NGRP, 16)
                       float* __restrict__ Hend)     // (NCHUNK, NGRP, 16)
{
    const int t    = blockIdx.x * blockDim.x + threadIdx.x;  // 0..524287
    const int n    = t & 15;
    const int gidx = t >> 4;                // c*2048 + bd
    const int bd   = gidx & (NGRP - 1);
    const int c    = gidx >> 11;
    const int b    = bd >> 10;
    const int d    = bd & 1023;

    const float An = Ap[d * DSTATE + n];
    const float Bn = Bp[d * DSTATE + n];
    const float w0 = conv_w[d * DCONV + 0];
    const float w1 = conv_w[d * DCONV + 1];
    const float w2 = conv_w[d * DCONV + 2];
    const float w3 = conv_w[d * DCONV + 3];
    const float cb = conv_b[d];

    const int l0 = c * LCHUNK;
    const float* xp = x     + ((size_t)b * SEQLEN + l0) * DMODEL + d;
    const float* dp = delta + ((size_t)b * SEQLEN + l0) * DMODEL + d;

    float x0 = 0.f, x1 = 0.f, x2 = 0.f;     // x[l-3], x[l-2], x[l-1]
    if (c > 0) {
        x0 = xp[-3 * (int)DMODEL];
        x1 = xp[-2 * (int)DMODEL];
        x2 = xp[-1 * (int)DMODEL];
    }
    float h = 0.f, P = 1.f;
    for (int l = 0; l < LCHUNK; ++l) {
        const float xl = xp[(size_t)l * DMODEL];
        const float dl = dp[(size_t)l * DMODEL];
        const float xc = w0 * x0 + w1 * x1 + w2 * x2 + w3 * xl + cb;
        x0 = x1; x1 = x2; x2 = xl;
        const float Ad = __expf(An * dl);
        P *= Ad;
        h = Ad * h + (Bn * dl) * xc;
    }
    const size_t o = (size_t)gidx * DSTATE + n;
    Pbuf[o] = P;
    Hend[o] = h;
}

__global__ __launch_bounds__(256)
void scan_part2_kernel(const float* __restrict__ Pbuf,
                       const float* __restrict__ Hend,
                       float* __restrict__ Hin)
{
    const int t  = blockIdx.x * blockDim.x + threadIdx.x;   // 0..32767
    const int n  = t & 15;
    const int bd = t >> 4;
    float h = 0.f;
    for (int c = 0; c < NCHUNK; ++c) {
        const size_t o = ((size_t)c * NGRP + bd) * DSTATE + n;
        Hin[o] = h;
        h = Pbuf[o] * h + Hend[o];
    }
}

__global__ __launch_bounds__(256)
void scan_part3_kernel(const float* __restrict__ x,
                       const float* __restrict__ Ap,
                       const float* __restrict__ Bp,
                       const float* __restrict__ Cp,
                       const float* __restrict__ Dp,
                       const float* __restrict__ conv_w,
                       const float* __restrict__ conv_b,
                       const float* __restrict__ delta,
                       const float* __restrict__ Hin,
                       float* __restrict__ y)
{
    const int t    = blockIdx.x * blockDim.x + threadIdx.x;
    const int n    = t & 15;
    const int gidx = t >> 4;
    const int bd   = gidx & (NGRP - 1);
    const int c    = gidx >> 11;
    const int b    = bd >> 10;
    const int d    = bd & 1023;

    const float An = Ap[d * DSTATE + n];
    const float Bn = Bp[d * DSTATE + n];
    const float Cn = Cp[d * DSTATE + n];
    const float Dd = Dp[d];
    const float w0 = conv_w[d * DCONV + 0];
    const float w1 = conv_w[d * DCONV + 1];
    const float w2 = conv_w[d * DCONV + 2];
    const float w3 = conv_w[d * DCONV + 3];
    const float cb = conv_b[d];

    const int l0 = c * LCHUNK;
    const float* xp = x     + ((size_t)b * SEQLEN + l0) * DMODEL + d;
    const float* dp = delta + ((size_t)b * SEQLEN + l0) * DMODEL + d;
    float*       yp = y     + ((size_t)b * SEQLEN + l0) * DMODEL + d;

    float x0 = 0.f, x1 = 0.f, x2 = 0.f;
    if (c > 0) {
        x0 = xp[-3 * (int)DMODEL];
        x1 = xp[-2 * (int)DMODEL];
        x2 = xp[-1 * (int)DMODEL];
    }
    float h = Hin[(size_t)gidx * DSTATE + n];

    for (int l = 0; l < LCHUNK; ++l) {
        const float xl = xp[(size_t)l * DMODEL];
        const float dl = dp[(size_t)l * DMODEL];
        const float xc = w0 * x0 + w1 * x1 + w2 * x2 + w3 * xl + cb;
        x0 = x1; x1 = x2; x2 = xl;
        const float Ad = __expf(An * dl);
        h = Ad * h + (Bn * dl) * xc;
        float yv = h * Cn;
        yv += __shfl_xor(yv, 1);
        yv += __shfl_xor(yv, 2);
        yv += __shfl_xor(yv, 4);
        yv += __shfl_xor(yv, 8);
        if (n == 0) yp[(size_t)l * DMODEL] = yv + Dd * xc;
    }
}

// ---------------------------------------------------------------------------
// Launch
// ---------------------------------------------------------------------------
extern "C" void kernel_launch(void* const* d_in, const int* in_sizes, int n_in,
                              void* d_out, int out_size, void* d_ws, size_t ws_size,
                              hipStream_t stream)
{
    const float* x      = (const float*)d_in[0];
    const float* A      = (const float*)d_in[1];
    const float* B      = (const float*)d_in[2];
    const float* C      = (const float*)d_in[3];
    const float* D      = (const float*)d_in[4];
    const float* Wd     = (const float*)d_in[5];
    const float* bd     = (const float*)d_in[6];
    const float* conv_w = (const float*)d_in[7];
    const float* conv_b = (const float*)d_in[8];

    // Workspace layout (26 MiB total):
    //   [ 0 MiB, 16 MiB)  delta (f32, 4096x1024)
    //   [16 MiB, 24 MiB)  x bf16   -- dead after GEMM, reused by P/Hend/Hin
    //   [24 MiB, 26 MiB)  Wd bf16
    char* ws = (char*)d_ws;
    float* delta = (float*)ws;
    char*  stage = ws + (size_t)M_TOTAL * N_DIM * sizeof(float);   // +16 MiB
    u16*   xbf   = (u16*)stage;
    u16*   wdbf  = (u16*)(stage + (size_t)M_TOTAL * K_DIM * sizeof(u16));
    float* Pbuf  = (float*)stage;                                   // 2 MiB
    float* Hend  = (float*)(stage + (2u << 20));                    // 2 MiB
    float* Hin   = (float*)(stage + (4u << 20));                    // 2 MiB

    cvt_bf16_kernel<<<1024, 256, 0, stream>>>(x, Wd, xbf, wdbf);
    gemm_delta_kernel<<<512, 256, 0, stream>>>(xbf, wdbf, bd, delta);
    scan_part1_kernel<<<2048, 256, 0, stream>>>(x, A, B, conv_w, conv_b,
                                                delta, Pbuf, Hend);
    scan_part2_kernel<<<128, 256, 0, stream>>>(Pbuf, Hend, Hin);
    scan_part3_kernel<<<2048, 256, 0, stream>>>(x, A, B, C, D, conv_w, conv_b,
                                                delta, Hin, (float*)d_out);
}
